// SpatialFuseModule_23072564314482
// MI455X (gfx1250) — matrix operand order repaired and checked by reading, not compiled, not run on hardware
//
#include <hip/hip_runtime.h>
#include <hip/hip_bf16.h>

#define CLS 20
#define NT  2304
#define NK  2324
#define CC  256
#define HDS 8
#define HD  32
#define BB  2
#define NPAD 2336
#define SCALE 0.17677669529663689f  // 32^-0.5

typedef _Float16 h16v __attribute__((ext_vector_type(16)));
typedef _Float16 h2v  __attribute__((ext_vector_type(2)));
typedef float    f8v  __attribute__((ext_vector_type(8)));

static __device__ __forceinline__ f8v wmma_f16(h16v a, h16v b, f8v c) {
  return __builtin_amdgcn_wmma_f32_16x16x32_f16(false, a, false, b, (short)0, c, false, false);
}

// A fragment 16x32 (MxK), row-major source. lane&15 = row, lane>>4 selects K-halves.
// Layout: lanes 0-15: K=0..15 ; lanes 16-31: K=16..31 (contiguous, matches B).
static __device__ __forceinline__ h16v load_a_rm(const _Float16* A, int ld, int row0, int k0,
                                                 int lane, int rowmax) {
  int l = lane & 15, hi = (lane >> 4) & 1;
  int row = row0 + l; row = row > rowmax ? rowmax : row;
  const _Float16* p = A + (long)row * ld + (k0 + hi * 16);
  h16v a;
#pragma unroll
  for (int j = 0; j < 4; ++j) {
    h2v t0 = *(const h2v*)(p + 2 * j);
    h2v t1 = *(const h2v*)(p + 8 + 2 * j);
    a[2 * j] = t0.x;     a[2 * j + 1] = t0.y;
    a[8 + 2 * j] = t1.x; a[8 + 2 * j + 1] = t1.y;
  }
  return a;
}

// B fragment 32x16 (KxN), source is K-major: element(k,n) at Bt[n*ld + k].
// Layout used here (contiguous, same as A): lane&15 = column n; half h -> K = (lane>>4)*16 + h.
static __device__ __forceinline__ h16v load_b_km(const _Float16* Bt, int ld, int col0, int k0,
                                                 int lane, int colmax) {
  int l = lane & 15, hi = (lane >> 4) & 1;
  int col = col0 + l; col = col > colmax ? colmax : col;
  const _Float16* p = Bt + (long)col * ld + (k0 + hi * 16);
  h16v b;
#pragma unroll
  for (int j = 0; j < 8; ++j) {
    h2v t = *(const h2v*)(p + 2 * j);
    b[2 * j] = t.x; b[2 * j + 1] = t.y;
  }
  return b;
}

// B fragment with per-pair K clamp (kmax even). Out-of-range K pairs read a valid
// address; the corresponding A (attn) values are zero so the product is exact.
static __device__ __forceinline__ h16v load_b_km_kclamp(const _Float16* Bt, int ld, int col0,
                                                        int k0, int lane, int kmax) {
  int l = lane & 15, hi = (lane >> 4) & 1;
  const _Float16* p = Bt + (long)(col0 + l) * ld;
  h16v b;
#pragma unroll
  for (int j = 0; j < 8; ++j) {
    int k = k0 + hi * 16 + 2 * j; k = k > kmax ? kmax : k;
    h2v t = *(const h2v*)(p + k);
    b[2 * j] = t.x; b[2 * j + 1] = t.y;
  }
  return b;
}

// ---------------- weight convert + transpose (f32 [K][N] -> f16 [N][K]) ----------------
__global__ void k_wt(const float* __restrict__ W, _Float16* __restrict__ Wt, int K, int Nn) {
  long i = (long)blockIdx.x * 256 + threadIdx.x;
  if (i >= (long)K * Nn) return;
  int k = (int)(i / Nn), n = (int)(i % Nn);
  Wt[(long)n * K + k] = (_Float16)W[i];
}

// ---------------- LayerNorm over C=256, one row per 256-thread block ----------------
__global__ void k_ln_q(const float* __restrict__ xin, const float* __restrict__ g,
                       const float* __restrict__ be, float* __restrict__ xq,
                       _Float16* __restrict__ qln) {
  int row = blockIdx.x, c = threadIdx.x;
  int b = row / NT, p = row - b * NT;
  float v = xin[((long)b * CC + c) * NT + p];   // NCHW gather
  __shared__ float sred[CC];
  sred[c] = v; __syncthreads();
  for (int s = 128; s > 0; s >>= 1) { if (c < s) sred[c] += sred[c + s]; __syncthreads(); }
  float mean = sred[0] * (1.f / CC);
  __syncthreads();
  float d = v - mean;
  sred[c] = d * d; __syncthreads();
  for (int s = 128; s > 0; s >>= 1) { if (c < s) sred[c] += sred[c + s]; __syncthreads(); }
  float var = sred[0] * (1.f / CC);
  float y = d * rsqrtf(var + 1e-5f) * g[c] + be[c];
  xq[(long)row * CC + c] = v;
  qln[(long)row * CC + c] = (_Float16)y;
}

__global__ void k_ln_rm(const float* __restrict__ xin, const float* __restrict__ g,
                        const float* __restrict__ be, _Float16* __restrict__ out) {
  int row = blockIdx.x, c = threadIdx.x;
  float v = xin[(long)row * CC + c];
  __shared__ float sred[CC];
  sred[c] = v; __syncthreads();
  for (int s = 128; s > 0; s >>= 1) { if (c < s) sred[c] += sred[c + s]; __syncthreads(); }
  float mean = sred[0] * (1.f / CC);
  __syncthreads();
  float d = v - mean;
  sred[c] = d * d; __syncthreads();
  for (int s = 128; s > 0; s >>= 1) { if (c < s) sred[c] += sred[c + s]; __syncthreads(); }
  float var = sred[0] * (1.f / CC);
  out[(long)row * CC + c] = (_Float16)(d * rsqrtf(var + 1e-5f) * g[c] + be[c]);
}

// ---------------- q = LN(xq) @ Wq  -> qheads[b][h][m][32] * SCALE (m >= CLS) ----------------
__global__ void k_qproj(const _Float16* __restrict__ qln, const _Float16* __restrict__ WqT,
                        _Float16* __restrict__ qheads) {
  int lane = threadIdx.x & 31, w = threadIdx.x >> 5;
  int b = blockIdx.x / (NT / 16);
  int m0 = (blockIdx.x % (NT / 16)) * 16;
  int col0 = blockIdx.y * 64 + w * 16;
  const _Float16* A = qln + (long)b * NT * CC;
  f8v c = {};
#pragma unroll
  for (int k = 0; k < CC; k += 32) {
    h16v a  = load_a_rm(A, CC, m0, k, lane, NT - 1);
    h16v bb = load_b_km(WqT, CC, col0, k, lane, CC - 1);
    c = wmma_f16(a, bb, c);
  }
  int l = lane & 15, hi = (lane >> 4) & 1;
  int col = col0 + l, hd = col >> 5, d = col & 31;
  _Float16* qp = qheads + ((long)b * HDS + hd) * NK * HD + d;
#pragma unroll
  for (int r = 0; r < 8; ++r) {
    int m = CLS + m0 + hi * 8 + r;
    qp[(long)m * HD] = (_Float16)(c[r] * SCALE);
  }
}

// ---------------- cls_t = LN(x_key)[:, :20] @ Wcls -> qheads rows m<CLS ----------------
__global__ void k_cls(const _Float16* __restrict__ kln, const _Float16* __restrict__ WclsT,
                      _Float16* __restrict__ qheads) {
  int lane = threadIdx.x & 31, w = threadIdx.x >> 5;
  int b = blockIdx.x >> 1;
  int m0 = (blockIdx.x & 1) * 16;
  int col0 = blockIdx.y * 64 + w * 16;
  const _Float16* A = kln + (long)b * NK * CC;
  f8v c = {};
  for (int k = 0; k < CC; k += 32) {
    h16v a  = load_a_rm(A, CC, m0, k, lane, CLS - 1);
    h16v bb = load_b_km(WclsT, CC, col0, k, lane, CC - 1);
    c = wmma_f16(a, bb, c);
  }
  int l = lane & 15, hi = (lane >> 4) & 1;
  int col = col0 + l, hd = col >> 5, d = col & 31;
  _Float16* qp = qheads + ((long)b * HDS + hd) * NK * HD + d;
  for (int r = 0; r < 8; ++r) {
    int m = m0 + hi * 8 + r;
    if (m < CLS) qp[(long)m * HD] = (_Float16)(c[r] * SCALE);
  }
}

// ---------------- kv = LN(x_key) @ Wkv -> kheads[b][h][n][32], vt[b][h][32][n] ----------------
__global__ void k_kv(const _Float16* __restrict__ kln, const _Float16* __restrict__ WkvT,
                     _Float16* __restrict__ kheads, _Float16* __restrict__ vt) {
  int lane = threadIdx.x & 31, w = threadIdx.x >> 5;
  const int tiles = (NK + 15) / 16;  // 146
  int b = blockIdx.x / tiles;
  int n0 = (blockIdx.x % tiles) * 16;
  int col0 = blockIdx.y * 64 + w * 16;
  const _Float16* A = kln + (long)b * NK * CC;
  f8v c = {};
  for (int k = 0; k < CC; k += 32) {
    h16v a  = load_a_rm(A, CC, n0, k, lane, NK - 1);
    h16v bb = load_b_km(WkvT, CC, col0, k, lane, 2 * CC - 1);
    c = wmma_f16(a, bb, c);
  }
  int l = lane & 15, hi = (lane >> 4) & 1;
  int col = col0 + l;
  for (int r = 0; r < 8; ++r) {
    int n = n0 + hi * 8 + r;
    if (n < NK) {
      if (col < CC) {
        int hd = col >> 5, d = col & 31;
        kheads[(((long)b * HDS + hd) * NK + n) * HD + d] = (_Float16)c[r];
      } else {
        int cv = col - CC, hd = cv >> 5, d = cv & 31;
        vt[(((long)b * HDS + hd) * HD + d) * NK + n] = (_Float16)c[r];
      }
    }
  }
}

// ---------------- fused attention: logits->LDS, split softmax, mask, softmax, attn@V ------------
__global__ void k_attn(const _Float16* __restrict__ qheads, const _Float16* __restrict__ kheads,
                       const _Float16* __restrict__ vt, const float* __restrict__ mask,
                       _Float16* __restrict__ ah) {
  extern __shared__ char smem[];
  _Float16* logit  = (_Float16*)smem;                    // 16*NPAD f16
  _Float16* attnh  = logit + 16 * NPAD;                  // 16*NPAD f16 (flag, then attn)
  float*    red    = (float*)(attnh + 16 * NPAD);        // 128
  float*    rowa   = red + 128;                          // 16
  float*    rowb   = rowa + 16;                          // 16
  float*    outacc = rowb + 16;                          // 16*32
  int tid = threadIdx.x, lane = tid & 31, w = tid >> 5;
  int bh = blockIdx.y, b = bh >> 3, hd = bh & 7;
  int m0 = blockIdx.x * 16;
  const _Float16* qb = qheads + (long)bh * NK * HD;
  const _Float16* kb = kheads + (long)bh * NK * HD;
  const _Float16* vb = vt + (long)bh * HD * NK;
  int l = lane & 15, hi = (lane >> 4) & 1;

  // logits (SCALE pre-folded into q); K=32 is one WMMA step
  h16v a = load_a_rm(qb, HD, m0, 0, lane, NK - 1);
  for (int t = w; t < NPAD / 16; t += 4) {
    int n0 = t * 16;
    h16v bb = load_b_km(kb, HD, n0, 0, lane, NK - 1);
    f8v c = {};
    c = wmma_f16(a, bb, c);
#pragma unroll
    for (int r = 0; r < 8; ++r) logit[(hi * 8 + r) * NPAD + n0 + l] = (_Float16)c[r];
  }
  __syncthreads();

  int rr = tid >> 3, sub = tid & 7;
  _Float16* Lr = logit + rr * NPAD;
  _Float16* Fr = attnh + rr * NPAD;

  // stage-1: split softmax over [0,CLS) and [CLS,NK)
  float mA = -1e30f, mB = -1e30f;
  for (int n = sub; n < CLS; n += 8) mA = fmaxf(mA, (float)Lr[n]);
  for (int n = CLS + sub; n < NK; n += 8) mB = fmaxf(mB, (float)Lr[n]);
  red[tid] = mA; __syncthreads();
  if (sub == 0) { float m = red[tid]; for (int j = 1; j < 8; ++j) m = fmaxf(m, red[tid + j]); rowa[rr] = m; }
  __syncthreads();
  red[tid] = mB; __syncthreads();
  if (sub == 0) { float m = red[tid]; for (int j = 1; j < 8; ++j) m = fmaxf(m, red[tid + j]); rowb[rr] = m; }
  __syncthreads();
  mA = rowa[rr]; mB = rowb[rr];
  float sA = 0.f, sB = 0.f;
  for (int n = sub; n < CLS; n += 8) sA += __expf((float)Lr[n] - mA);
  for (int n = CLS + sub; n < NK; n += 8) sB += __expf((float)Lr[n] - mB);
  red[tid] = sA; __syncthreads();
  if (sub == 0) { float s = 0.f; for (int j = 0; j < 8; ++j) s += red[tid + j]; rowa[rr] = s; }
  __syncthreads();
  red[tid] = sB; __syncthreads();
  if (sub == 0) { float s = 0.f; for (int j = 0; j < 8; ++j) s += red[tid + j]; rowb[rr] = s; }
  __syncthreads();
  float rA = 1.f / rowa[rr], rB = 1.f / rowb[rr];
  for (int n = sub; n < NK; n += 8) {
    float v = (float)Lr[n];
    Lr[n] = (_Float16)((n < CLS) ? __expf(v - mA) * rA : __expf(v - mB) * rB);
  }
  __syncthreads();

  // stage-2: mask read ONCE (345MB stream), bern flag cached in Fr, then softmax
  int mrow = m0 + rr; if (mrow > NK - 1) mrow = NK - 1;
  const float* mp = mask + ((long)bh * NK + mrow) * NK;
  float m2 = -1e30f;
  for (int n = sub; n < NK; n += 8) {
    bool msk = mp[n] < 0.3f;
    Fr[n] = msk ? (_Float16)1.f : (_Float16)0.f;
    if (!msk) m2 = fmaxf(m2, (float)Lr[n]);
  }
  red[tid] = m2; __syncthreads();
  if (sub == 0) { float m = red[tid]; for (int j = 1; j < 8; ++j) m = fmaxf(m, red[tid + j]); rowa[rr] = m; }
  __syncthreads();
  m2 = rowa[rr];
  float s2 = 0.f;
  for (int n = sub; n < NK; n += 8)
    if ((float)Fr[n] == 0.f) s2 += __expf((float)Lr[n] - m2);
  red[tid] = s2; __syncthreads();
  if (sub == 0) { float s = 0.f; for (int j = 0; j < 8; ++j) s += red[tid + j]; rowb[rr] = s; }
  __syncthreads();
  s2 = rowb[rr];
  float r2 = s2 > 0.f ? 1.f / s2 : 0.f;
  for (int n = sub; n < NPAD; n += 8) {   // also zeroes the K-pad columns
    float o = 0.f;
    if (n < NK && (float)Fr[n] == 0.f) o = __expf((float)Lr[n] - m2) * r2;
    Fr[n] = (_Float16)o;
  }
  for (int i = tid; i < 512; i += 128) outacc[i] = 0.f;
  __syncthreads();

  // attn @ V : 4-wave K-split, ds_add_f32 cross-wave reduction
  f8v c0 = {}, c1 = {};
  for (int kk = w * 32; kk < NPAD; kk += 128) {
    h16v aa = load_a_rm(attnh, NPAD, 0, kk, lane, 15);
    h16v b0 = load_b_km_kclamp(vb, NK, 0,  kk, lane, NK - 2);
    h16v b1 = load_b_km_kclamp(vb, NK, 16, kk, lane, NK - 2);
    c0 = wmma_f16(aa, b0, c0);
    c1 = wmma_f16(aa, b1, c1);
  }
#pragma unroll
  for (int r = 0; r < 8; ++r) {
    int row = hi * 8 + r;
    atomicAdd(&outacc[row * 32 + l], c0[r]);
    atomicAdd(&outacc[row * 32 + 16 + l], c1[r]);
  }
  __syncthreads();
  for (int i = tid; i < 512; i += 128) {
    int row = i >> 5, d = i & 31, m = m0 + row;
    if (m < NK) ah[((long)b * NK + m) * CC + hd * HD + d] = (_Float16)outacc[i];
  }
}

// ---------------- x = attn_out[:, CLS:] @ Wp + bp + xq ----------------
__global__ void k_proj(const _Float16* __restrict__ ah, const _Float16* __restrict__ WpT,
                       const float* __restrict__ bp, const float* __restrict__ xq,
                       float* __restrict__ xr) {
  int lane = threadIdx.x & 31, w = threadIdx.x >> 5;
  int b = blockIdx.x / (NT / 16);
  int p0 = (blockIdx.x % (NT / 16)) * 16;
  int col0 = blockIdx.y * 64 + w * 16;
  const _Float16* A = ah + ((long)b * NK + CLS) * CC;
  f8v c = {};
  for (int k = 0; k < CC; k += 32) {
    h16v a  = load_a_rm(A, CC, p0, k, lane, NT - 1);
    h16v bb = load_b_km(WpT, CC, col0, k, lane, CC - 1);
    c = wmma_f16(a, bb, c);
  }
  int l = lane & 15, hi = (lane >> 4) & 1;
  int col = col0 + l;
  float bias = bp[col];
  for (int r = 0; r < 8; ++r) {
    long idx = ((long)b * NT + p0 + hi * 8 + r) * CC + col;
    xr[idx] = xq[idx] + c[r] + bias;
  }
}

// ---------------- h1 = gelu(LN3(xr) @ W1 + bf1) ----------------
__global__ void k_mlp1(const _Float16* __restrict__ hln, const _Float16* __restrict__ W1T,
                       const float* __restrict__ bf1, _Float16* __restrict__ h1) {
  int lane = threadIdx.x & 31, w = threadIdx.x >> 5;
  int r0 = blockIdx.x * 16;
  int col0 = blockIdx.y * 64 + w * 16;
  f8v c = {};
  for (int k = 0; k < CC; k += 32) {
    __builtin_prefetch(hln + (long)r0 * CC + k + 64, 0, 1);
    h16v a  = load_a_rm(hln, CC, r0, k, lane, BB * NT - 1);
    h16v bb = load_b_km(W1T, CC, col0, k, lane, 4 * CC - 1);
    c = wmma_f16(a, bb, c);
  }
  int l = lane & 15, hi = (lane >> 4) & 1;
  int col = col0 + l;
  float bias = bf1[col];
  for (int r = 0; r < 8; ++r) {
    float v = c[r] + bias;
    float g = 0.5f * v * (1.f + erff(v * 0.70710678f));   // exact GELU
    h1[(long)(r0 + hi * 8 + r) * (4 * CC) + col] = (_Float16)g;
  }
}

// ---------------- out = (xr + h1 @ W2 + bf2) transposed back to NCHW ----------------
__global__ void k_mlp2(const _Float16* __restrict__ h1, const _Float16* __restrict__ W2T,
                       const float* __restrict__ bf2, const float* __restrict__ xr,
                       float* __restrict__ out) {
  int lane = threadIdx.x & 31, w = threadIdx.x >> 5;
  int r0 = blockIdx.x * 16;
  int col0 = blockIdx.y * 64 + w * 16;
  f8v c = {};
  for (int k = 0; k < 4 * CC; k += 32) {
    __builtin_prefetch(h1 + (long)r0 * (4 * CC) + k + 64, 0, 1);
    h16v a  = load_a_rm(h1, 4 * CC, r0, k, lane, BB * NT - 1);
    h16v bb = load_b_km(W2T, 4 * CC, col0, k, lane, CC - 1);
    c = wmma_f16(a, bb, c);
  }
  int l = lane & 15, hi = (lane >> 4) & 1;
  int col = col0 + l;
  float bias = bf2[col];
  for (int r = 0; r < 8; ++r) {
    int row = r0 + hi * 8 + r;
    int b = row / NT, p = row - b * NT;
    float o = xr[(long)row * CC + col] + c[r] + bias;
    out[((long)b * CC + col) * NT + p] = o;
  }
}

extern "C" void kernel_launch(void* const* d_in, const int* in_sizes, int n_in,
                              void* d_out, int out_size, void* d_ws, size_t ws_size,
                              hipStream_t stream) {
  (void)in_sizes; (void)n_in; (void)out_size; (void)ws_size;
  const float* x_query = (const float*)d_in[0];
  const float* x_key   = (const float*)d_in[1];
  const float* mask_u  = (const float*)d_in[2];
  const float* g1  = (const float*)d_in[3];
  const float* be1 = (const float*)d_in[4];
  const float* g2  = (const float*)d_in[5];
  const float* be2 = (const float*)d_in[6];
  const float* g3  = (const float*)d_in[7];
  const float* be3 = (const float*)d_in[8];
  const float* Wq  = (const float*)d_in[9];
  const float* Wkv = (const float*)d_in[10];
  const float* Wcls= (const float*)d_in[11];
  const float* Wp  = (const float*)d_in[12];
  const float* bp  = (const float*)d_in[13];
  const float* W1  = (const float*)d_in[14];
  const float* bf1 = (const float*)d_in[15];
  const float* W2  = (const float*)d_in[16];
  const float* bf2 = (const float*)d_in[17];

  char* ws = (char*)d_ws; size_t off = 0;
  auto alloc = [&](size_t bytes) { void* p = ws + off; off = (off + bytes + 255) & ~(size_t)255; return p; };
  _Float16* WqT   = (_Float16*)alloc((size_t)256 * 256 * 2);
  _Float16* WclsT = (_Float16*)alloc((size_t)256 * 256 * 2);
  _Float16* WkvT  = (_Float16*)alloc((size_t)256 * 512 * 2);
  _Float16* WpT   = (_Float16*)alloc((size_t)256 * 256 * 2);
  _Float16* W1T   = (_Float16*)alloc((size_t)256 * 1024 * 2);
  _Float16* W2T   = (_Float16*)alloc((size_t)1024 * 256 * 2);
  _Float16* qln   = (_Float16*)alloc((size_t)BB * NT * CC * 2);
  _Float16* kln   = (_Float16*)alloc((size_t)BB * NK * CC * 2);
  float*    xqf   = (float*)   alloc((size_t)BB * NT * CC * 4);
  _Float16* qheads= (_Float16*)alloc((size_t)BB * HDS * NK * HD * 2);
  _Float16* kheads= (_Float16*)alloc((size_t)BB * HDS * NK * HD * 2);
  _Float16* vt    = (_Float16*)alloc((size_t)BB * HDS * HD * NK * 2);
  _Float16* ah    = (_Float16*)alloc((size_t)BB * NK * CC * 2);
  float*    xr    = (float*)   alloc((size_t)BB * NT * CC * 4);
  _Float16* hln   = (_Float16*)alloc((size_t)BB * NT * CC * 2);
  _Float16* h1    = (_Float16*)alloc((size_t)BB * NT * 4 * CC * 2);

  k_wt<<<(256 * 256 + 255) / 256, 256, 0, stream>>>(Wq, WqT, 256, 256);
  k_wt<<<(256 * 256 + 255) / 256, 256, 0, stream>>>(Wcls, WclsT, 256, 256);
  k_wt<<<(256 * 512 + 255) / 256, 256, 0, stream>>>(Wkv, WkvT, 256, 512);
  k_wt<<<(256 * 256 + 255) / 256, 256, 0, stream>>>(Wp, WpT, 256, 256);
  k_wt<<<(256 * 1024 + 255) / 256, 256, 0, stream>>>(W1, W1T, 256, 1024);
  k_wt<<<(1024 * 256 + 255) / 256, 256, 0, stream>>>(W2, W2T, 1024, 256);

  k_ln_q<<<BB * NT, 256, 0, stream>>>(x_query, g1, be1, xqf, qln);
  k_ln_rm<<<BB * NK, 256, 0, stream>>>(x_key, g2, be2, kln);

  k_qproj<<<dim3(BB * (NT / 16), CC / 64), 128, 0, stream>>>(qln, WqT, qheads);
  k_cls<<<dim3(BB * 2, CC / 64), 128, 0, stream>>>(kln, WclsT, qheads);
  k_kv<<<dim3(BB * ((NK + 15) / 16), (2 * CC) / 64), 128, 0, stream>>>(kln, WkvT, kheads, vt);

  size_t smem = (size_t)2 * 16 * NPAD * 2 + (size_t)(128 + 16 + 16 + 512) * 4;  // ~152 KB (<320 KB WGP LDS)
  k_attn<<<dim3((NK + 15) / 16, BB * HDS), 128, smem, stream>>>(qheads, kheads, vt, mask_u, ah);

  k_proj<<<dim3(BB * (NT / 16), CC / 64), 128, 0, stream>>>(ah, WpT, bp, xqf, xr);
  k_ln_rm<<<BB * NT, 256, 0, stream>>>(xr, g3, be3, hln);
  k_mlp1<<<dim3(BB * NT / 16, (4 * CC) / 64), 128, 0, stream>>>(hln, W1T, bf1, h1);
  k_mlp2<<<dim3(BB * NT / 16, CC / 64), 128, 0, stream>>>(h1, W2T, bf2, xr, (float*)d_out);
}
